// StellarByteMoEGate_5970004541879
// MI455X (gfx1250) — matrix operand order repaired and checked by reading, not compiled, not run on hardware
//
#include <hip/hip_runtime.h>
#include <hip/hip_bf16.h>
#include <math.h>

// ---------------------------------------------------------------------------
// MoE gate for MI455X (gfx1250, wave32, WMMA).
//   logits[T,64] = x[T,2048] @ W[64,2048]^T  via v_wmma_f32_16x16x32_bf16
//   softmax + top-2 + renorm + seq-aux loss fused (x is read exactly once).
// ---------------------------------------------------------------------------

typedef __attribute__((ext_vector_type(16))) __bf16 v16bf;
typedef __attribute__((ext_vector_type(8)))  float  v8f;

#define BSZ   4
#define SEQ   4096
#define TOKS  (BSZ * SEQ)     // 16384
#define HDIM  2048
#define NEXP  64
#define TOPK  2
#define ALPHA 0.01f

#define WAVES_PER_BLOCK  8
#define TOKENS_PER_WAVE  16
#define TOKENS_PER_BLOCK (WAVES_PER_BLOCK * TOKENS_PER_WAVE)  // 128

__global__ void sbmoe_init_ws(float* ws) {
    ws[threadIdx.x] = 0.0f;   // 512 floats: ce[4][64] + score_sum[4][64]
}

__global__ __launch_bounds__(256)
void sbmoe_gate_kernel(const float* __restrict__ x,
                       const float* __restrict__ w,
                       float* __restrict__ out_idx,   // [T,2] (stored as float)
                       float* __restrict__ out_wgt,   // [T,2]
                       float* __restrict__ ws_ce,     // [4,64]
                       float* __restrict__ ws_ss)     // [4,64]
{
    __shared__ float lds_logits[WAVES_PER_BLOCK][TOKENS_PER_WAVE][NEXP]; // 32 KB
    __shared__ float blk_ce[NEXP];
    __shared__ float blk_ss[NEXP];

    const int tid  = threadIdx.x;
    const int wave = tid >> 5;
    const int lane = tid & 31;
    const int m    = lane & 15;   // row within 16 (token / expert-in-tile)
    const int h    = lane >> 4;   // lane half

    if (tid < NEXP) { blk_ce[tid] = 0.0f; blk_ss[tid] = 0.0f; }
    __syncthreads();

    const int token_base = blockIdx.x * TOKENS_PER_BLOCK + wave * TOKENS_PER_WAVE;
    const float* xrow = x + (size_t)(token_base + m) * HDIM;

    v8f acc0 = {}, acc1 = {}, acc2 = {}, acc3 = {};

    for (int k0 = 0; k0 < HDIM; k0 += 32) {
        __builtin_prefetch(xrow + k0 + 256, 0, 1);  // global_prefetch_b8

        // ---- A fragment: 16x32 bf16 (contiguous k = 16h + s, same order as B) ----
        // elems 0..7  <- K = k0 + 16h + 0..7
        // elems 8..15 <- K = k0 + 16h + 8 + 0..7
        const float4* pa0 = (const float4*)(xrow + k0 + 16 * h);
        const float4* pa1 = (const float4*)(xrow + k0 + 16 * h + 8);
        float4 a0 = pa0[0], a1 = pa0[1];
        float4 a2 = pa1[0], a3 = pa1[1];
        float af[16] = {a0.x,a0.y,a0.z,a0.w, a1.x,a1.y,a1.z,a1.w,
                        a2.x,a2.y,a2.z,a2.w, a3.x,a3.y,a3.z,a3.w};
        v16bf A;
        #pragma unroll
        for (int j = 0; j < 16; ++j) A[j] = (__bf16)af[j];

        // ---- 4 B fragments (expert tiles), L2-resident weight ----
        #pragma unroll
        for (int t = 0; t < 4; ++t) {
            // B 32x16: lane n = m holds K = k0 + 16h + 0..15 of expert 16t+n
            const float4* pb = (const float4*)(w + (size_t)(16 * t + m) * HDIM
                                               + k0 + 16 * h);
            float4 b0 = pb[0], b1 = pb[1], b2 = pb[2], b3 = pb[3];
            float bf[16] = {b0.x,b0.y,b0.z,b0.w, b1.x,b1.y,b1.z,b1.w,
                            b2.x,b2.y,b2.z,b2.w, b3.x,b3.y,b3.z,b3.w};
            v16bf B;
            #pragma unroll
            for (int j = 0; j < 16; ++j) B[j] = (__bf16)bf[j];

            if      (t == 0) acc0 = __builtin_amdgcn_wmma_f32_16x16x32_bf16(false, A, false, B, (short)0, acc0, false, false);
            else if (t == 1) acc1 = __builtin_amdgcn_wmma_f32_16x16x32_bf16(false, A, false, B, (short)0, acc1, false, false);
            else if (t == 2) acc2 = __builtin_amdgcn_wmma_f32_16x16x32_bf16(false, A, false, B, (short)0, acc2, false, false);
            else             acc3 = __builtin_amdgcn_wmma_f32_16x16x32_bf16(false, A, false, B, (short)0, acc3, false, false);
        }
    }

    // ---- spill logits to wave-private LDS; C/D layout: M = r + 8h, N = m ----
    float* myl = &lds_logits[wave][0][0];
    #pragma unroll
    for (int r = 0; r < 8; ++r) {
        const int tok = r + 8 * h;
        myl[tok * NEXP +  0 + m] = acc0[r];
        myl[tok * NEXP + 16 + m] = acc1[r];
        myl[tok * NEXP + 32 + m] = acc2[r];
        myl[tok * NEXP + 48 + m] = acc3[r];
    }
    asm volatile("s_wait_dscnt 0" ::: "memory");   // same-wave LDS RAW

    // ---- softmax + top-2 + aux accumulation: lanes 0..15, one token each ----
    if (h == 0) {
        const float* lv = &lds_logits[wave][m][0];
        float mx = -INFINITY;
        #pragma unroll 8
        for (int e = 0; e < NEXP; ++e) mx = fmaxf(mx, lv[e]);

        float sum = 0.0f;
        float b1 = -INFINITY, b2 = -INFINITY;
        int   i1 = 0, i2 = 0;
        for (int e = 0; e < NEXP; ++e) {
            float v = lv[e];
            sum += __expf(v - mx);
            if (v > b1)      { b2 = b1; i2 = i1; b1 = v; i1 = e; }
            else if (v > b2) { b2 = v; i2 = e; }
        }
        const float inv = 1.0f / sum;
        const float w1  = __expf(b1 - mx) * inv;
        const float w2  = __expf(b2 - mx) * inv;
        const float dn  = 1.0f / (w1 + w2 + 1e-20f);

        const int tok = token_base + m;
        out_idx[2 * tok + 0] = (float)i1;
        out_idx[2 * tok + 1] = (float)i2;
        out_wgt[2 * tok + 0] = w1 * dn;
        out_wgt[2 * tok + 1] = w2 * dn;

        atomicAdd(&blk_ce[i1], 1.0f);
        atomicAdd(&blk_ce[i2], 1.0f);
        for (int e = 0; e < NEXP; ++e)
            atomicAdd(&blk_ss[e], __expf(lv[e] - mx) * inv);
    }
    __syncthreads();

    // one global atomic per (batch, expert) per block
    if (tid < NEXP) {
        const int b = (blockIdx.x * TOKENS_PER_BLOCK) / SEQ;  // uniform per block
        atomicAdd(&ws_ce[b * NEXP + tid], blk_ce[tid]);
        atomicAdd(&ws_ss[b * NEXP + tid], blk_ss[tid]);
    }
}

__global__ void sbmoe_aux_kernel(const float* __restrict__ ws_ce,
                                 const float* __restrict__ ws_ss,
                                 float* __restrict__ out_aux)
{
    __shared__ float red[BSZ * NEXP];
    const int tid = threadIdx.x;          // 256 = BSZ*NEXP
    // ce_norm = count / (S*K/E);  mean_scores = ss / S
    const float ce = ws_ce[tid] * ((float)NEXP / (float)(SEQ * TOPK));
    const float ms = ws_ss[tid] / (float)SEQ;
    red[tid] = ce * ms;
    __syncthreads();
    for (int s = 128; s > 0; s >>= 1) {
        if (tid < s) red[tid] += red[tid + s];
        __syncthreads();
    }
    if (tid == 0) out_aux[0] = red[0] * (ALPHA / (float)BSZ);
}

extern "C" void kernel_launch(void* const* d_in, const int* in_sizes, int n_in,
                              void* d_out, int out_size, void* d_ws, size_t ws_size,
                              hipStream_t stream)
{
    const float* x = (const float*)d_in[0];   // [4,4096,2048] fp32
    const float* w = (const float*)d_in[1];   // [64,2048]     fp32

    float* out   = (float*)d_out;
    float* o_idx = out;                 // [T,2] topk_idx (as float)
    float* o_wgt = out + 2 * TOKS;      // [T,2] topk_weight
    float* o_aux = out + 4 * TOKS;      // [1]   aux_loss

    float* ws    = (float*)d_ws;
    float* ws_ce = ws;                  // [4,64]
    float* ws_ss = ws + BSZ * NEXP;     // [4,64]

    sbmoe_init_ws<<<1, 2 * BSZ * NEXP, 0, stream>>>(ws);
    sbmoe_gate_kernel<<<TOKS / TOKENS_PER_BLOCK, 256, 0, stream>>>(
        x, w, o_idx, o_wgt, ws_ce, ws_ss);
    sbmoe_aux_kernel<<<1, BSZ * NEXP, 0, stream>>>(ws_ce, ws_ss, o_aux);
}
